// IBertIntermediate_55336358641926
// MI455X (gfx1250) — 8-bit matrix operand order repaired and checked by reading, compiled, not run on hardware
//
#include <hip/hip_runtime.h>
#include <math.h>


typedef int v8i __attribute__((ext_vector_type(8)));
typedef int gv4i __attribute__((vector_size(16)));

#define TILE_M 128
#define TILE_N 128
#define TILE_K 64

#if defined(__has_builtin)
#if __has_builtin(__builtin_amdgcn_global_load_async_to_lds_b128) && \
    __has_builtin(__builtin_amdgcn_s_wait_asynccnt)
#define USE_ASYNC_LDS 1
#endif
#endif

#define AS1G(p) ((__attribute__((address_space(1))) gv4i*)(p))
#define AS3L(p) ((__attribute__((address_space(3))) gv4i*)(p))

__device__ __forceinline__ void stage16(const signed char* g, signed char* l) {
#ifdef USE_ASYNC_LDS
  __builtin_amdgcn_global_load_async_to_lds_b128(AS1G(g), AS3L(l), 0, 0);
#else
  *(uint4*)l = *(const uint4*)g;
#endif
}

__device__ __forceinline__ unsigned fkey(float f) {
  unsigned u = __float_as_uint(f);
  return (u & 0x80000000u) ? ~u : (u | 0x80000000u);
}
__device__ __forceinline__ float funkey(unsigned k) {
  unsigned u = (k & 0x80000000u) ? (k ^ 0x80000000u) : ~k;
  return __uint_as_float(u);
}

__global__ void k_init(unsigned* minmax) {
  minmax[0] = fkey(0.0f);
  minmax[1] = fkey(0.0f);
}

__global__ void k_wquant(const float* __restrict__ W, const float* __restrict__ bias,
                         const float* __restrict__ sfp, int H,
                         signed char* __restrict__ wq,
                         float* __restrict__ biasq, float* __restrict__ bq,
                         float* __restrict__ cq, float* __restrict__ shiftv,
                         float* __restrict__ osf) {
  const int row = blockIdx.x;
  const int tid = threadIdx.x;
  const float* wrow = W + (size_t)row * H;

  float mn = 3.4e38f, mx = -3.4e38f;
  for (int j = tid; j < H; j += blockDim.x) {
    float v = wrow[j];
    mn = fminf(mn, v);
    mx = fmaxf(mx, v);
  }
  __shared__ float smn[256], smx[256];
  __shared__ float s_fc;
  smn[tid] = mn; smx[tid] = mx;
  __syncthreads();
  for (int s = 128; s > 0; s >>= 1) {
    if (tid < s) {
      smn[tid] = fminf(smn[tid], smn[tid + s]);
      smx[tid] = fmaxf(smx[tid], smx[tid + s]);
    }
    __syncthreads();
  }
  if (tid == 0) {
    float sat = fmaxf(fabsf(smn[0]), fabsf(smx[0]));
    float fc = fmaxf(sat, 1e-8f) / 127.0f;
    s_fc = fc;
    float prev = sfp[0];
    float bsf  = fc * prev;
    biasq[row] = rintf(bias[row] / bsf);
    float sfe = bsf / 1.4142f;
    bq[row] = floorf(-1.769f / sfe);
    cq[row] = floorf((1.0f / -0.2888f) / (sfe * sfe));
    float sig = sfe * sfe * (-0.2888f) * 16384.0f;
    shiftv[row] = floorf(1.0f / sig);
    osf[row] = bsf * sig * 0.5f;
  }
  __syncthreads();
  const float fc = s_fc;
  for (int j = tid; j < H; j += blockDim.x) {
    float q = rintf(wrow[j] / fc);
    q = fminf(fmaxf(q, -127.0f), 126.0f);
    wq[(size_t)row * H + j] = (signed char)(int)q;
  }
}

__global__ void k_xquant(const float* __restrict__ x, const float* __restrict__ sfp,
                         signed char* __restrict__ xq, int n) {
  int i = blockIdx.x * blockDim.x + threadIdx.x;
  if (i < n) {
    float q = rintf(x[i] / sfp[0]);
    q = fminf(fmaxf(q, -127.0f), 126.0f);
    xq[i] = (signed char)(int)q;
  }
}

__global__ void __launch_bounds__(256, 2)
k_gemm_gelu(const signed char* __restrict__ xq, const signed char* __restrict__ wq,
            int K, int N,
            const float* __restrict__ biasq, const float* __restrict__ bq,
            const float* __restrict__ cq, const float* __restrict__ shiftv,
            const float* __restrict__ osf,
            float* __restrict__ z, unsigned* __restrict__ minmax) {
  __shared__ __align__(16) signed char As[2][TILE_M * TILE_K];
  __shared__ __align__(16) signed char Bs[2][TILE_N * TILE_K];
  __shared__ unsigned s_mn, s_mx;

  const int tid   = threadIdx.x;
  const int lane  = tid & 31;
  const int wid   = tid >> 5;
  const int waveM = wid >> 2;
  const int waveN = wid & 3;
  const int lm    = lane & 15;
  const int hi    = lane >> 4;

  const int mBase = blockIdx.x * TILE_M;
  const int nBase = blockIdx.y * TILE_N;

  if (tid == 0) { s_mn = 0xFFFFFFFFu; s_mx = 0u; }

  const int c0 = tid,         r0 = c0 >> 2, k0 = (c0 & 3) * 16;
  const int c1 = tid + 256,   r1 = c1 >> 2, k1 = (c1 & 3) * 16;
  const signed char* gA0 = xq + (size_t)(mBase + r0) * K + k0;
  const signed char* gA1 = xq + (size_t)(mBase + r1) * K + k1;
  const signed char* gB0 = wq + (size_t)(nBase + r0) * K + k0;
  const signed char* gB1 = wq + (size_t)(nBase + r1) * K + k1;
  const int lA0 = r0 * TILE_K + k0;
  const int lA1 = r1 * TILE_K + k1;

  v8i acc[4][2];
#pragma unroll
  for (int mt = 0; mt < 4; ++mt)
#pragma unroll
    for (int nt = 0; nt < 2; ++nt)
#pragma unroll
      for (int r = 0; r < 8; ++r) acc[mt][nt][r] = 0;

  stage16(gA0, As[0] + lA0);
  stage16(gA1, As[0] + lA1);
  stage16(gB0, Bs[0] + lA0);
  stage16(gB1, Bs[0] + lA1);
  gA0 += TILE_K; gA1 += TILE_K; gB0 += TILE_K; gB1 += TILE_K;

  const int nK = K / TILE_K;
  for (int i = 0; i < nK; ++i) {
#ifdef USE_ASYNC_LDS
    __builtin_amdgcn_s_wait_asynccnt(0);
#endif
    __syncthreads();

    if (i + 1 < nK) {
      const int b = (i + 1) & 1;
      stage16(gA0, As[b] + lA0);
      stage16(gA1, As[b] + lA1);
      stage16(gB0, Bs[b] + lA0);
      stage16(gB1, Bs[b] + lA1);
      gA0 += TILE_K; gA1 += TILE_K; gB0 += TILE_K; gB1 += TILE_K;
    }

    const signed char* Ab = As[i & 1];
    const signed char* Bb = Bs[i & 1];

    union { v8i v; uint4 q[2]; } bf[2];
#pragma unroll
    for (int nt = 0; nt < 2; ++nt) {
      const signed char* bp = Bb + (waveN * 32 + nt * 16 + lm) * TILE_K + hi * 16;
      bf[nt].q[0] = *(const uint4*)(bp);
      bf[nt].q[1] = *(const uint4*)(bp + 32);
    }
#pragma unroll
    for (int mt = 0; mt < 4; ++mt) {
      union { v8i v; unsigned long long u[4]; } af;
      const unsigned long long* ap =
          (const unsigned long long*)(Ab + (waveM * 64 + mt * 16 + lm) * TILE_K + hi * 16);
      af.u[0] = ap[0]; af.u[1] = ap[1]; af.u[2] = ap[4]; af.u[3] = ap[5];
#pragma unroll
      for (int nt = 0; nt < 2; ++nt) {
        acc[mt][nt] = __builtin_amdgcn_wmma_i32_16x16x64_iu8(
            true, af.v, true, bf[nt].v, acc[mt][nt], false, false);
      }
    }
  }

  float lmn = 3.4e38f, lmx = -3.4e38f;
#pragma unroll
  for (int nt = 0; nt < 2; ++nt) {
    const int ch = nBase + waveN * 32 + nt * 16 + lm;
    const float bqv = biasq[ch];
    const float b_i = bq[ch];
    const float c_i = cq[ch];
    const float sh  = shiftv[ch];
    const float os  = osf[ch];
#pragma unroll
    for (int mt = 0; mt < 4; ++mt) {
#pragma unroll
      for (int r = 0; r < 8; ++r) {
        float xi  = (float)acc[mt][nt][r] + bqv;
        float sgn = (xi > 0.0f) ? 1.0f : ((xi < 0.0f) ? -1.0f : 0.0f);
        float ai  = fminf(fabsf(xi), -b_i);
        float y   = sgn * ((ai + b_i) * (ai + b_i) + c_i);
        float yi  = floorf(y * 6.103515625e-05f);
        float t   = xi * (yi + sh);
        float h   = t * os;
        lmn = fminf(lmn, h);
        lmx = fmaxf(lmx, h);
        const int m = mBase + waveM * 64 + mt * 16 + hi * 8 + r;
        z[(size_t)m * N + ch] = t;
      }
    }
  }
  atomicMin(&s_mn, fkey(lmn));
  atomicMax(&s_mx, fkey(lmx));
  __syncthreads();
  if (tid == 0) {
    atomicMin(&minmax[0], s_mn);
    atomicMax(&minmax[1], s_mx);
  }
}

__global__ void k_factors(const unsigned* __restrict__ minmax,
                          const float* __restrict__ osf,
                          float* __restrict__ factor, float* __restrict__ sf_out,
                          int N) {
  int ch = blockIdx.x * blockDim.x + threadIdx.x;
  if (ch >= N) return;
  float mn = fminf(funkey(minmax[0]), 0.0f);
  float mx = fmaxf(funkey(minmax[1]), 0.0f);
  float act_sf = fmaxf(fmaxf(fabsf(mn), fabsf(mx)), 1e-8f) / 127.0f;
  int e;
  float m = frexpf(osf[ch] / act_sf, &e);
  float m_int = floorf(m * 2147483648.0f + 0.5f);
  factor[ch] = m_int * exp2f((float)(e - 31));
  if (ch == 0) sf_out[0] = act_sf;
}

__global__ void k_requant(float* __restrict__ z, const float* __restrict__ factor,
                          const unsigned* __restrict__ minmax, int N, size_t total) {
  size_t i = (size_t)blockIdx.x * blockDim.x + threadIdx.x;
  if (i >= total) return;
  float mn = fminf(funkey(minmax[0]), 0.0f);
  float mx = fmaxf(funkey(minmax[1]), 0.0f);
  float act_sf = fmaxf(fmaxf(fabsf(mn), fabsf(mx)), 1e-8f) / 127.0f;
  int ch = (int)(i % (size_t)N);
  float t = rintf(z[i]);
  float v = rintf(t * factor[ch]);
  v = fminf(fmaxf(v, -128.0f), 127.0f);
  z[i] = v * act_sf;
}

extern "C" void kernel_launch(void* const* d_in, const int* in_sizes, int n_in,
                              void* d_out, int out_size, void* d_ws, size_t ws_size,
                              hipStream_t stream) {
  const float* x    = (const float*)d_in[0];
  const float* sfp  = (const float*)d_in[1];
  const float* W    = (const float*)d_in[2];
  const float* bias = (const float*)d_in[3];

  const int I = in_sizes[3];
  const int H = in_sizes[2] / I;
  const int M = in_sizes[0] / H;

  char* ws = (char*)d_ws;
  unsigned* minmax = (unsigned*)ws;
  float* biasq  = (float*)(ws + 256);
  float* bq     = biasq + I;
  float* cq     = bq + I;
  float* shiftv = cq + I;
  float* osf    = shiftv + I;
  float* factor = osf + I;
  size_t off = 256 + (size_t)6 * I * sizeof(float);
  off = (off + 255) & ~(size_t)255;
  signed char* wq = (signed char*)(ws + off);
  off += (size_t)I * H;
  off = (off + 255) & ~(size_t)255;
  signed char* xq = (signed char*)(ws + off);

  float* z      = (float*)d_out;
  float* sf_out = z + (size_t)M * I;

  k_init<<<1, 1, 0, stream>>>(minmax);
  k_wquant<<<I, 256, 0, stream>>>(W, bias, sfp, H, wq, biasq, bq, cq, shiftv, osf);
  {
    int n = M * H;
    k_xquant<<<(n + 255) / 256, 256, 0, stream>>>(x, sfp, xq, n);
  }
  dim3 grid(M / TILE_M, I / TILE_N);
  k_gemm_gelu<<<grid, 256, 0, stream>>>(xq, wq, H, I, biasq, bq, cq, shiftv, osf,
                                        z, minmax);
  k_factors<<<(I + 255) / 256, 256, 0, stream>>>(minmax, osf, factor, sf_out, I);
  {
    size_t total = (size_t)M * I;
    k_requant<<<(unsigned)((total + 255) / 256), 256, 0, stream>>>(z, factor, minmax,
                                                                   I, total);
  }
}
